// EGraphSAGE_76029511073825
// MI455X (gfx1250) — matrix operand order repaired and checked by reading, not compiled, not run on hardware
//
#include <hip/hip_runtime.h>

#define N_NODES 100000
#define N_EDGES 1600000

typedef __attribute__((ext_vector_type(16))) __bf16 v16bf;
typedef __attribute__((ext_vector_type(8)))  __bf16 v8bf;
typedef __attribute__((ext_vector_type(8)))  float  v8f;

// ---------------------------------------------------------------------------
// helpers
// ---------------------------------------------------------------------------
static __device__ __forceinline__ v16bf cat16(v8bf lo, v8bf hi) {
    v16bf r;
#pragma unroll
    for (int i = 0; i < 8; ++i) { r[i] = lo[i]; r[i + 8] = hi[i]; }
    return r;
}

static __device__ __forceinline__ v8bf cvt8(const float* __restrict__ p, float scale) {
    v8bf r;
#pragma unroll
    for (int i = 0; i < 8; ++i) r[i] = (__bf16)(p[i] * scale);
    return r;
}

// ---------------------------------------------------------------------------
// fp32 -> bf16 bulk convert
// ---------------------------------------------------------------------------
__global__ void cvt_bf16_kernel(const float* __restrict__ in, __bf16* __restrict__ out, long n) {
    long i = (long)blockIdx.x * blockDim.x + threadIdx.x;
    if (i < n) out[i] = (__bf16)in[i];
}

// ---------------------------------------------------------------------------
// Pack W [64 x Kd] (row major, fan_out x fan_in) into WMMA B fragments:
// out[kc][nt][lane][i] = W[nt*16 + lane%16][kc*32 + (i&7) + 8*(lane>=16) + 16*(i>=8)]
// ---------------------------------------------------------------------------
__global__ void pack_weights_kernel(const float* __restrict__ W, __bf16* __restrict__ out,
                                    int kchunks, int Kd) {
    int idx = blockIdx.x * blockDim.x + threadIdx.x;
    int total = kchunks * 4 * 32 * 16;
    if (idx >= total) return;
    int i  = idx & 15;
    int l  = (idx >> 4) & 31;
    int nt = (idx >> 9) & 3;
    int kc = idx >> 11;
    int k  = kc * 32 + (i & 7) + ((l >= 16) ? 8 : 0) + ((i >= 8) ? 16 : 0);
    int n  = nt * 16 + (l & 15);
    out[idx] = (__bf16)W[n * Kd + k];
}

// ---------------------------------------------------------------------------
// in-degree counts (same for both layers: dst indices don't change)
// ---------------------------------------------------------------------------
__global__ void count_kernel(const int* __restrict__ ei, float* __restrict__ cnt) {
    long e = (long)blockIdx.x * blockDim.x + threadIdx.x;
    if (e < N_EDGES) atomicAdd(cnt + ei[N_EDGES + e], 1.0f);
}

// ---------------------------------------------------------------------------
// Edge message GEMM + scatter-add.
// Wave32 owns 16 edges. A = [gather(feat[src]) | cvt(edge_attr)] : 16x96 bf16.
// B = packed W_msg^T (96x64), staged in LDS once per block (12 KB).
// D(16x64, f32) scatter-atomicAdd'ed to sums[dst].
// ---------------------------------------------------------------------------
__global__ __launch_bounds__(256)
void edge_msg_kernel(const __bf16* __restrict__ feat,   // [N_NODES,64] bf16
                     const float*  __restrict__ ea,     // [N_EDGES,32] f32
                     const int*    __restrict__ ei,     // [2,N_EDGES]
                     const __bf16* __restrict__ Wpk,    // [3][4][32][16] bf16
                     const float*  __restrict__ bias,   // [64]
                     float*        __restrict__ sums)   // [N_NODES,64]
{
    __shared__ __align__(32) __bf16 sB[3 * 4 * 32 * 16];   // 12288 B

    const int tid  = threadIdx.x;
    const int wave = (blockIdx.x * blockDim.x + tid) >> 5;
    const int lane = tid & 31;
    const int e0   = wave * 16;
    const int m    = lane & 15;
    const int hi   = lane >> 4;

    // ---- issue per-wave gathers (in flight across the barrier) ----------
    const int src = ei[e0 + m];
    long rowbase[8];
#pragma unroll
    for (int v = 0; v < 8; ++v)
        rowbase[v] = (long)ei[N_EDGES + e0 + v + hi * 8] * 64;

    const __bf16* xrow = feat + (long)src * 64;
    const float*  erow = ea   + (long)(e0 + m) * 32;

    v16bf afrag[3];
#pragma unroll
    for (int kc = 0; kc < 2; ++kc) {
        const __bf16* p = xrow + kc * 32 + hi * 8;
        afrag[kc] = cat16(*(const v8bf*)p, *(const v8bf*)(p + 16));
    }
    {
        const float* p = erow + hi * 8;
        afrag[2] = cat16(cvt8(p, 1.0f), cvt8(p + 16, 1.0f));
    }

    // ---- cooperative global -> LDS weight stage (all 256 threads) -------
    {
        const uint4* g = (const uint4*)Wpk;
        uint4* s = (uint4*)sB;
#pragma unroll
        for (int it = 0; it < 3; ++it)            // 768 uint4 / 256 threads
            s[it * 256 + tid] = g[it * 256 + tid];
    }
    __syncthreads();

    v8f acc[4];
#pragma unroll
    for (int nt = 0; nt < 4; ++nt) {
        float b = bias[nt * 16 + m];
#pragma unroll
        for (int v = 0; v < 8; ++v) acc[nt][v] = b;
    }

    // ---- 12 WMMAs, B fed from LDS ---------------------------------------
    const v16bf* Bmat = (const v16bf*)sB;
#pragma unroll
    for (int kc = 0; kc < 3; ++kc)
#pragma unroll
        for (int nt = 0; nt < 4; ++nt)
            acc[nt] = __builtin_amdgcn_wmma_f32_16x16x32_bf16(
                false, afrag[kc], false, Bmat[(kc * 4 + nt) * 32 + lane],
                (short)0, acc[nt], false, false);

    // ---- scatter-add ----------------------------------------------------
#pragma unroll
    for (int nt = 0; nt < 4; ++nt) {
        int col = nt * 16 + m;
#pragma unroll
        for (int v = 0; v < 8; ++v)
            atomicAdd(sums + rowbase[v] + col, acc[nt][v]);
    }
}

// ---------------------------------------------------------------------------
// Node apply: A = [self_feats | sums/max(cnt,1)] : 16x128 bf16,
// B = packed W_apply^T (128x64) staged in LDS. out = relu(A@B+b) bf16.
// ---------------------------------------------------------------------------
__global__ __launch_bounds__(256)
void node_apply_kernel(const __bf16* __restrict__ self,  // [N_NODES,64] bf16
                       const float*  __restrict__ sums,  // [N_NODES,64]
                       const float*  __restrict__ cnt,   // [N_NODES]
                       const __bf16* __restrict__ Wpk,   // [4][4][32][16]
                       const float*  __restrict__ bias,  // [64]
                       __bf16*       __restrict__ out)   // [N_NODES,64]
{
    __shared__ __align__(32) __bf16 sB[4 * 4 * 32 * 16];   // 16384 B

    const int tid  = threadIdx.x;
    const int wave = (blockIdx.x * blockDim.x + tid) >> 5;
    const int lane = tid & 31;
    const int n0   = wave * 16;
    const int m    = lane & 15;
    const int hi   = lane >> 4;

    // ---- cooperative global -> LDS weight stage (all waves, incl. tail) --
    {
        const uint4* g = (const uint4*)Wpk;
        uint4* s = (uint4*)sB;
#pragma unroll
        for (int it = 0; it < 4; ++it)            // 1024 uint4 / 256 threads
            s[it * 256 + tid] = g[it * 256 + tid];
    }
    __syncthreads();

    if (n0 >= N_NODES) return;                     // wave-uniform tail guard

    const int node = n0 + m;
    const __bf16* srow = self + (long)node * 64;
    const float*  arow = sums + (long)node * 64;
    const float inv = 1.0f / fmaxf(cnt[node], 1.0f);

    v16bf afrag[4];
#pragma unroll
    for (int kc = 0; kc < 2; ++kc) {
        const __bf16* p = srow + kc * 32 + hi * 8;
        afrag[kc] = cat16(*(const v8bf*)p, *(const v8bf*)(p + 16));
    }
#pragma unroll
    for (int kc = 2; kc < 4; ++kc) {
        const float* p = arow + (kc - 2) * 32 + hi * 8;
        afrag[kc] = cat16(cvt8(p, inv), cvt8(p + 16, inv));
    }

    v8f acc[4];
#pragma unroll
    for (int nt = 0; nt < 4; ++nt) {
        float b = bias[nt * 16 + m];
#pragma unroll
        for (int v = 0; v < 8; ++v) acc[nt][v] = b;
    }

    const v16bf* Bmat = (const v16bf*)sB;
#pragma unroll
    for (int kc = 0; kc < 4; ++kc)
#pragma unroll
        for (int nt = 0; nt < 4; ++nt)
            acc[nt] = __builtin_amdgcn_wmma_f32_16x16x32_bf16(
                false, afrag[kc], false, Bmat[(kc * 4 + nt) * 32 + lane],
                (short)0, acc[nt], false, false);

#pragma unroll
    for (int v = 0; v < 8; ++v) {
        long rb = (long)(n0 + v + hi * 8) * 64;
#pragma unroll
        for (int nt = 0; nt < 4; ++nt)
            out[rb + nt * 16 + m] = (__bf16)fmaxf(acc[nt][v], 0.0f);
    }
}

// ---------------------------------------------------------------------------
// Edge predictor: out[e,:] = concat(z[row], z[col]) @ Wp^T + bp   (OUT=2)
// ---------------------------------------------------------------------------
__global__ __launch_bounds__(256)
void edge_pred_kernel(const __bf16* __restrict__ z,  // [N_NODES,64] bf16
                      const int*    __restrict__ ei, // [2,N_EDGES]
                      const float*  __restrict__ Wp, // [2,128]
                      const float*  __restrict__ bp, // [2]
                      float*        __restrict__ out) // [N_EDGES,2]
{
    __shared__ float sW[256];
    int tid = threadIdx.x;
    sW[tid] = Wp[tid];
    __syncthreads();
    long e = (long)blockIdx.x * blockDim.x + tid;
    if (e >= N_EDGES) return;
    int r = ei[e], c = ei[N_EDGES + e];
    const __bf16* zr = z + (long)r * 64;
    const __bf16* zc = z + (long)c * 64;
    float a0 = bp[0], a1 = bp[1];
#pragma unroll
    for (int k = 0; k < 64; ++k) {
        float f = (float)zr[k];
        a0 += f * sW[k];
        a1 += f * sW[128 + k];
    }
#pragma unroll
    for (int k = 0; k < 64; ++k) {
        float f = (float)zc[k];
        a0 += f * sW[64 + k];
        a1 += f * sW[192 + k];
    }
    out[e * 2 + 0] = a0;
    out[e * 2 + 1] = a1;
}

// ---------------------------------------------------------------------------
// launch
// ---------------------------------------------------------------------------
extern "C" void kernel_launch(void* const* d_in, const int* in_sizes, int n_in,
                              void* d_out, int out_size, void* d_ws, size_t ws_size,
                              hipStream_t stream) {
    (void)in_sizes; (void)n_in; (void)out_size; (void)ws_size;
    const float* x        = (const float*)d_in[0];
    const float* ea       = (const float*)d_in[1];
    const float* W1_msg   = (const float*)d_in[2];
    const float* b1_msg   = (const float*)d_in[3];
    const float* W1_apply = (const float*)d_in[4];
    const float* b1_apply = (const float*)d_in[5];
    const float* W2_msg   = (const float*)d_in[6];
    const float* b2_msg   = (const float*)d_in[7];
    const float* W2_apply = (const float*)d_in[8];
    const float* b2_apply = (const float*)d_in[9];
    const float* Wp       = (const float*)d_in[10];
    const float* bp       = (const float*)d_in[11];
    const int*   ei       = (const int*)d_in[12];
    float* out = (float*)d_out;

    char* ws = (char*)d_ws;
    size_t off = 0;
    auto alloc = [&](size_t bytes) {
        void* p = ws + off;
        off += (bytes + 255) & ~(size_t)255;
        return p;
    };
    __bf16* xb    = (__bf16*)alloc((size_t)N_NODES * 64 * 2);
    __bf16* hb    = (__bf16*)alloc((size_t)N_NODES * 64 * 2);
    __bf16* zb    = (__bf16*)alloc((size_t)N_NODES * 64 * 2);
    float*  sums  = (float*) alloc((size_t)N_NODES * 64 * 4);
    float*  cnt   = (float*) alloc((size_t)N_NODES * 4);
    __bf16* Wm1p  = (__bf16*)alloc(3 * 4 * 32 * 16 * 2);
    __bf16* Wm2p  = (__bf16*)alloc(3 * 4 * 32 * 16 * 2);
    __bf16* Wa1p  = (__bf16*)alloc(4 * 4 * 32 * 16 * 2);
    __bf16* Wa2p  = (__bf16*)alloc(4 * 4 * 32 * 16 * 2);

    // prep: x -> bf16, pack all weights into WMMA B-fragment layout
    {
        long n = (long)N_NODES * 64;
        cvt_bf16_kernel<<<(int)((n + 255) / 256), 256, 0, stream>>>(x, xb, n);
    }
    pack_weights_kernel<<<(3 * 4 * 32 * 16 + 255) / 256, 256, 0, stream>>>(W1_msg, Wm1p, 3, 96);
    pack_weights_kernel<<<(3 * 4 * 32 * 16 + 255) / 256, 256, 0, stream>>>(W2_msg, Wm2p, 3, 96);
    pack_weights_kernel<<<(4 * 4 * 32 * 16 + 255) / 256, 256, 0, stream>>>(W1_apply, Wa1p, 4, 128);
    pack_weights_kernel<<<(4 * 4 * 32 * 16 + 255) / 256, 256, 0, stream>>>(W2_apply, Wa2p, 4, 128);

    hipMemsetAsync(sums, 0, (size_t)N_NODES * 64 * 4, stream);
    hipMemsetAsync(cnt,  0, (size_t)N_NODES * 4, stream);
    count_kernel<<<(N_EDGES + 255) / 256, 256, 0, stream>>>(ei, cnt);

    const int edgeBlocks = (N_EDGES / 16) * 32 / 256;                 // 12500 (exact)
    const int nodeBlocks = ((N_NODES + 15) / 16 * 32 + 255) / 256;    // 782

    // layer 1
    edge_msg_kernel<<<edgeBlocks, 256, 0, stream>>>(xb, ea, ei, Wm1p, b1_msg, sums);
    node_apply_kernel<<<nodeBlocks, 256, 0, stream>>>(xb, sums, cnt, Wa1p, b1_apply, hb);

    // layer 2
    hipMemsetAsync(sums, 0, (size_t)N_NODES * 64 * 4, stream);
    edge_msg_kernel<<<edgeBlocks, 256, 0, stream>>>(hb, ea, ei, Wm2p, b2_msg, sums);
    node_apply_kernel<<<nodeBlocks, 256, 0, stream>>>(hb, sums, cnt, Wa2p, b2_apply, zb);

    // edge predictor
    edge_pred_kernel<<<(N_EDGES + 255) / 256, 256, 0, stream>>>(zb, ei, Wp, bp, out);
}
